// Lstm_27925877359075
// MI455X (gfx1250) — matrix operand order repaired and checked by reading, not compiled, not run on hardware
//
#include <hip/hip_runtime.h>
#include <hip/hip_bf16.h>

#define EMBED 300
#define HID   300
#define GP    304           // gate width padded to 19*16
#define NPAD  1216          // 4*GP
#define KT    10            // K padded 300->320, 10 k-tiles of 32
#define TT    256
#define BB    64
#define NT    76            // NPAD/16 n-tiles
#define MT_X  1024          // (T*B)/16 m-tiles for the big GEMM
#define HASZ  (4*KT*32*16)  // packed h buffer elements (bf16) = 20480

typedef __attribute__((ext_vector_type(16))) __bf16 v16bf;
typedef __attribute__((ext_vector_type(8)))  float  v8f;

__device__ __forceinline__ float sigmoidf_(float x){ return 1.0f/(1.0f+__expf(-x)); }

// ---------------------------------------------------------------------------
// Kernel 1: pack Wx/Wh into WMMA-B bf16 layout, pad bias, zero h ping-pong,
// reset the grid-barrier counter. B layout: N = lane&15, K = kt*32+16*(e>>3)+8*(lane>>4)+(e&7)
// ---------------------------------------------------------------------------
__global__ void prep_kernel(const float* __restrict__ W, const float* __restrict__ b,
                            __bf16* __restrict__ WxP, __bf16* __restrict__ WhP,
                            float* __restrict__ bP, __bf16* __restrict__ hAg,
                            unsigned int* __restrict__ ctr)
{
  int flat = blockIdx.x*256 + threadIdx.x;
  const int WCNT = 2*NT*KT*32;          // 48640 packing threads
  const int HCNT = 2*HASZ;              // 40960 zeroing threads
  if (flat < WCNT) {
    int lane  = flat & 31;
    int kt    = (flat>>5) % KT;
    int nt    = (flat/(32*KT)) % NT;
    int which = flat/(32*KT*NT);        // 0 = Wx (rows 0..299), 1 = Wh (rows 300..599)
    int base  = which ? EMBED : 0;
    __bf16* dst = (which ? WhP : WxP) + ((size_t)(nt*KT + kt)*32 + lane)*16;
    int N  = lane & 15;
    int np = nt*16 + N;
    int g  = np / GP;
    int wi = np % GP;
    #pragma unroll
    for (int e=0;e<16;e++){
      int K = kt*32 + ((e>>3)<<4) + ((lane>>4)<<3) + (e&7);
      float v = (K < HID && wi < HID) ? W[(size_t)(base+K)*(4*HID) + g*HID + wi] : 0.0f;
      dst[e] = (__bf16)v;
    }
    return;
  }
  flat -= WCNT;
  if (flat < HCNT) { hAg[flat] = (__bf16)0.0f; return; }
  flat -= HCNT;
  if (flat < NPAD) {
    int g = flat / GP, wi = flat % GP;
    bP[flat] = (wi < HID) ? b[g*HID + wi] : 0.0f;
    return;
  }
  flat -= NPAD;
  if (flat == 0) *ctr = 0u;
}

// ---------------------------------------------------------------------------
// Kernel 2: gather emb rows for all (t,b), convert to bf16, pack into WMMA-A
// layout. A layout: M = lane&15, K = kt*32 + 16*(e>>3) + 8*(lane>>4) + (e&7)
// ---------------------------------------------------------------------------
__global__ void pack_x_kernel(const float* __restrict__ emb, const int* __restrict__ ids,
                              __bf16* __restrict__ XP)
{
  int flat = blockIdx.x*256 + threadIdx.x;     // [mt][kt][lane]
  if (flat >= MT_X*KT*32) return;
  int lane = flat & 31;
  int kt   = (flat>>5) % KT;
  int mt   = flat/(32*KT);
  int m    = mt*16 + (lane & 15);              // row m = t*64 + batch
  int t    = m >> 6;
  int brow = m & 63;
  int id   = ids[brow*TT + t];
  const float* row = emb + (size_t)id*EMBED;
  __bf16* dst = XP + (size_t)flat*16;
  #pragma unroll
  for (int e=0;e<16;e++){
    int K = kt*32 + ((e>>3)<<4) + ((lane>>4)<<3) + (e&7);
    dst[e] = (__bf16)((K < EMBED) ? row[K] : 0.0f);
  }
}

// ---------------------------------------------------------------------------
// Kernel 3: big GEMM  XG = X @ Wx + b  (16384 x 1216, K=320) via WMMA bf16.
// One 16x16 tile per wave; results stored in C-layout tile order + bias.
// ---------------------------------------------------------------------------
__global__ void gemm_x_kernel(const __bf16* __restrict__ XP, const __bf16* __restrict__ WxP,
                              const float* __restrict__ bP, float* __restrict__ XG)
{
  int wave = threadIdx.x >> 5;
  int lane = threadIdx.x & 31;
  int tile = blockIdx.x*8 + wave;
  if (tile >= MT_X*NT) return;
  int mt = tile / NT;
  int nt = tile % NT;
  v8f acc = {};
  const v16bf* Abase = (const v16bf*)XP  + ((size_t)mt*KT*32 + lane);
  const v16bf* Bbase = (const v16bf*)WxP + ((size_t)nt*KT*32 + lane);
  #pragma unroll
  for (int kt=0;kt<KT;kt++){
    v16bf a  = Abase[kt*32];
    v16bf bv = Bbase[kt*32];
    acc = __builtin_amdgcn_wmma_f32_16x16x32_bf16(false, a, false, bv, (short)0, acc, false, false);
  }
  float bias = bP[nt*16 + (lane & 15)];
  #pragma unroll
  for (int r=0;r<8;r++) acc[r] += bias;
  *(v8f*)(XG + ((size_t)tile*32 + lane)*8) = acc;
}

// ---------------------------------------------------------------------------
// Kernel 4: recurrent scan. 19 WGs (one per q column) x 4 waves (one per
// 16-row M tile). Cell state c lives in registers for all 256 steps; h
// ping-pongs through a 40KB WMMA-A-packed global buffer (L2 resident),
// staged into LDS each step via async memory->LDS DMA (ASYNCcnt).
// Device-scope release/acquire atomic barrier between steps.
// ---------------------------------------------------------------------------
__global__ void __launch_bounds__(128) lstm_kernel(const float* __restrict__ XG,
                                                   const __bf16* __restrict__ WhP,
                                                   const int* __restrict__ lens,
                                                   __bf16* __restrict__ hAg,
                                                   unsigned int* __restrict__ ctr)
{
  __shared__ __bf16 hA[HASZ];                 // 40KB staged copy of packed h
  const int tid  = threadIdx.x;
  const int lane = tid & 31;
  const int mt   = tid >> 5;                  // wave id == M tile
  const int q    = blockIdx.x;                // 0..18

  int lens8[8];
  #pragma unroll
  for (int r=0;r<8;r++) lens8[r] = lens[mt*16 + r + ((lane>>4)<<3)];

  v8f c = {};                                 // cell state, register-resident

  // LDS byte offset of hA (low 32 bits of the generic LDS pointer)
  const unsigned int ldsbase = (unsigned int)(unsigned long long)(&hA[0]);

  for (int t=0;t<TT;t++){
    const int rp = t & 1;
    // stage hAg[rp] -> LDS via async DMA: 2560 x 16B chunks over 128 threads
    {
      const char* src = (const char*)(hAg + (size_t)rp*HASZ);
      #pragma unroll
      for (int i=0;i<20;i++){
        unsigned int doff = ldsbase + (unsigned int)((i*128 + tid)*16);
        const char*  s    = src + (size_t)(i*128 + tid)*16;
        asm volatile("global_load_async_to_lds_b128 %0, %1, off"
                     :: "v"(doff), "v"(s) : "memory");
      }
      asm volatile("s_wait_asynccnt 0x0" ::: "memory");
    }
    __syncthreads();

    // prefetch next step's accumulator tiles while we do WMMA work
    if (t+1 < TT) {
      #pragma unroll
      for (int g=0; g<4; g++){
        int nt  = g*19 + q;
        int mtg = (t+1)*4 + mt;
        __builtin_prefetch(XG + ((size_t)(mtg*NT + nt)*32 + lane)*8, 0, 1);
      }
    }

    v8f acc[4];
    #pragma unroll
    for (int g=0; g<4; g++){
      int nt  = g*19 + q;
      int mtg = t*4 + mt;
      acc[g] = *(const v8f*)(XG + ((size_t)(mtg*NT + nt)*32 + lane)*8);
      #pragma unroll
      for (int kt=0;kt<KT;kt++){
        v16bf a  = *(const v16bf*)(hA  + ((size_t)(mt*KT + kt)*32 + lane)*16);
        v16bf bv = *(const v16bf*)(WhP + (((size_t)(nt*KT + kt))*32 + lane)*16);
        acc[g] = __builtin_amdgcn_wmma_f32_16x16x32_bf16(false, a, false, bv, (short)0, acc[g], false, false);
      }
    }

    // elementwise: i,j,f,o are local to this wave's 16x16 tile
    __bf16* hout = hAg + (size_t)(1-rp)*HASZ;
    #pragma unroll
    for (int r=0;r<8;r++){
      float ig = sigmoidf_(acc[0][r]);
      float jg = tanhf(acc[1][r]);
      float fg = sigmoidf_(acc[2][r] + 1.0f);   // FORGET_BIAS
      float og = sigmoidf_(acc[3][r]);
      float cn = fg*c[r] + ig*jg;
      float hn = og*tanhf(cn);
      int idx   = q*16 + (lane & 15);           // hidden index 0..303
      int kk    = idx & 31, ktA = idx >> 5;
      int laneA = (r + ((lane>>4)<<3)) + (((kk>>3)&1)<<4);
      int e     = ((kk>>4)<<3) + (kk&7);
      int off   = ((mt*KT + ktA)*32 + laneA)*16 + e;
      bool upd  = (t < lens8[r]);
      float hold = (float)hA[off];              // previous h (for masked rows)
      c[r] = upd ? cn : c[r];
      hout[off] = (__bf16)(upd ? hn : hold);
    }

    // grid barrier: release all h stores, wait for all 19 WGs
    __threadfence();
    __syncthreads();
    if (tid == 0) __hip_atomic_fetch_add(ctr, 1u, __ATOMIC_ACQ_REL, __HIP_MEMORY_SCOPE_AGENT);
    const unsigned int target = 19u*(unsigned)(t+1);
    while (__hip_atomic_load(ctr, __ATOMIC_ACQUIRE, __HIP_MEMORY_SCOPE_AGENT) < target)
      __builtin_amdgcn_s_sleep(1);
  }
}

// ---------------------------------------------------------------------------
// Kernel 5: logits = h @ Wd + bd (64x3). Final h is in ping-pong buffer 0.
// ---------------------------------------------------------------------------
__global__ void logits_kernel(const __bf16* __restrict__ hAg, const float* __restrict__ Wd,
                              const float* __restrict__ bd, float* __restrict__ out)
{
  int tid = threadIdx.x;
  if (tid >= BB*3) return;
  int row = tid/3, cls = tid%3;
  int mt = row>>4, Mr = row&15;
  float s = bd[cls];
  for (int k=0;k<HID;k++){
    int kk = k & 31, kt = k >> 5;
    int laneA = Mr + (((kk>>3)&1)<<4);
    int e = ((kk>>4)<<3) + (kk&7);
    float h = (float)hAg[((mt*KT + kt)*32 + laneA)*16 + e];
    s += h * Wd[k*3 + cls];
  }
  out[row*3 + cls] = s;
}

extern "C" void kernel_launch(void* const* d_in, const int* in_sizes, int n_in,
                              void* d_out, int out_size, void* d_ws, size_t ws_size,
                              hipStream_t stream)
{
  const float* emb = (const float*)d_in[0];
  const float* W   = (const float*)d_in[1];
  const float* b   = (const float*)d_in[2];
  const float* Wd  = (const float*)d_in[3];
  const float* bd  = (const float*)d_in[4];
  const int*   ids = (const int*)d_in[5];
  const int*   lens= (const int*)d_in[6];
  float* out = (float*)d_out;

  char* ws = (char*)d_ws;
  size_t off = 0;
  auto alloc = [&](size_t bytes)->void* {
    void* p = ws + off;
    off = (off + bytes + 255) & ~(size_t)255;
    return p;
  };
  __bf16* XP  = (__bf16*)alloc((size_t)MT_X*KT*32*16*2);   // 10.5 MB packed x
  __bf16* WxP = (__bf16*)alloc((size_t)NT*KT*32*16*2);     // 760 KB
  __bf16* WhP = (__bf16*)alloc((size_t)NT*KT*32*16*2);     // 760 KB
  float*  bP  = (float*)alloc((size_t)NPAD*4);
  float*  XG  = (float*)alloc((size_t)MT_X*NT*32*8*4);     // 76 MB xg+bias tiles
  __bf16* hAg = (__bf16*)alloc((size_t)2*HASZ*2);          // 80 KB h ping-pong
  unsigned int* ctr = (unsigned int*)alloc(64);

  prep_kernel  <<<355, 256, 0, stream>>>(W, b, WxP, WhP, bP, hAg, ctr);
  pack_x_kernel<<<1280,256, 0, stream>>>(emb, ids, XP);
  gemm_x_kernel<<<9728,256, 0, stream>>>(XP, WxP, bP, XG);
  lstm_kernel  <<<19,  128, 0, stream>>>(XG, WhP, lens, hAg, ctr);
  logits_kernel<<<1,   256, 0, stream>>>(hAg, Wd, bd, out);
}
